// GraphTransformerNetLayer_19095424598407
// MI455X (gfx1250) — matrix operand order repaired and checked by reading, not compiled, not run on hardware
//
#include <hip/hip_runtime.h>
#include <hip/hip_bf16.h>
#include <math.h>

#define N_NODES 50000
#define N_EDGES 800000
#define HID     128
#define HEADS   4
#define LAYERS  2
#define LN_EPS  1e-5f

typedef __attribute__((ext_vector_type(16))) _Float16 v16h;
typedef __attribute__((ext_vector_type(8)))  _Float16 v8h;
typedef __attribute__((ext_vector_type(8)))  float    v8f;

// nan_to_num(nan=0, posinf=0, neginf=0)
__device__ __forceinline__ float finz(float v) {
    return (v == v && v != INFINITY && v != -INFINITY) ? v : 0.0f;
}
// order-preserving float<->uint encoding for atomicMax-based segment max
__device__ __forceinline__ unsigned fenc(float f) {
    unsigned u = __float_as_uint(f);
    return (u & 0x80000000u) ? ~u : (u | 0x80000000u);
}
__device__ __forceinline__ float fdec(unsigned u) {
    return __uint_as_float((u & 0x80000000u) ? (u & 0x7fffffffu) : ~u);
}

// ---------------------------------------------------------------------------
// Pack per-layer [Wq|Wk|Wv|Wskip] (each [128,128]) into WMMA-B fragment order:
// wfrag[((l*4+kb)*32+ct)*32*16 + lane*16 + e] = Wcat[K = kb*32+16*(lane/16)+e]
//                                                   [N = ct*16 + lane%16]
// Also pack concatenated bias bcat[l][512].
// ---------------------------------------------------------------------------
__global__ void prep_weights(const float* __restrict__ Wq, const float* __restrict__ Wk,
                             const float* __restrict__ Wv, const float* __restrict__ Ws,
                             const float* __restrict__ bq, const float* __restrict__ bk,
                             const float* __restrict__ bv, const float* __restrict__ bs,
                             _Float16* __restrict__ wfrag, float* __restrict__ bcat) {
    int t = blockIdx.x * 256 + threadIdx.x;
    if (t < LAYERS * 4 * 32 * 32 * 16) {
        int e  = t & 15;
        int ln = (t >> 4)  & 31;
        int ct = (t >> 9)  & 31;
        int kb = (t >> 14) & 3;
        int l  = t >> 16;
        int K   = kb * 32 + (ln >> 4) * 16 + e;
        int col = ct * 16 + (ln & 15);
        int sel = col >> 7, c = col & 127;
        const float* W = (sel == 0) ? Wq : (sel == 1) ? Wk : (sel == 2) ? Wv : Ws;
        wfrag[t] = (_Float16)W[((size_t)l * HID + K) * HID + c];
    }
    if (t < LAYERS * 512) {
        int l = t >> 9, col = t & 511;
        int sel = col >> 7, c = col & 127;
        const float* B = (sel == 0) ? bq : (sel == 1) ? bk : (sel == 2) ? bv : bs;
        bcat[t] = B[l * HID + c];
    }
}

// Clean x and produce both f32 (residual path) and f16 (WMMA A path) copies.
__global__ void copy_clean(const float* __restrict__ x, float* __restrict__ h,
                           _Float16* __restrict__ h16, int n) {
    int i = blockIdx.x * 256 + threadIdx.x;
    if (i < n) {
        float v = finz(x[i]);
        h[i] = v;
        h16[i] = (_Float16)v;
    }
}

__global__ void init_layer(float* __restrict__ accum, float* __restrict__ nsum,
                           unsigned* __restrict__ nmax) {
    int i = blockIdx.x * 256 + threadIdx.x;
    if (i < N_NODES * HID) accum[i] = 0.0f;
    if (i < N_NODES * HEADS) { nsum[i] = 0.0f; nmax[i] = 0u; }
}

// ---------------------------------------------------------------------------
// Fused GEMM: OUT[n, 0:512] = h[n, :] @ [Wq|Wk|Wv|Wskip] + bias
// Grid: (3125 row-tiles, 4 col-groups). Block: 256 thr = 8 waves, wave w owns
// 16x16 tile at columns (blockIdx.y*8+w)*16. K=128 -> 4x v_wmma_f32_16x16x32_f16.
// A tile staged straight from the persistent f16 activation copy into LDS via
// the gfx1250 async memory->LDS path (no VGPR round-trip), ASYNCcnt tracked.
// ---------------------------------------------------------------------------
__global__ void __launch_bounds__(256, 2)
gemm_qkvs(const _Float16* __restrict__ h16, const _Float16* __restrict__ wfrag,
          const float* __restrict__ bcat, float* __restrict__ qkvs) {
    __shared__ _Float16 Ash[16 * 136];   // 16 rows x 128 K, stride 136 to skew banks
    const int tid = threadIdx.x;
    const int rowbase = blockIdx.x * 16;

    // Stage A tile (16 rows x 128 halves = 4KB), 16 bytes per thread,
    // memory -> LDS directly (GLOBAL_LOAD_ASYNC_TO_LDS_B128).
    {
        int row = tid >> 4;      // 0..15
        int seg = tid & 15;      // 16-byte segment within row
        const _Float16* gsrc = h16 + (size_t)(rowbase + row) * HID + seg * 8;
        _Float16* ldst = &Ash[row * 136 + seg * 8];
        unsigned lds_off = (unsigned)(uintptr_t)ldst;  // low 32 bits = LDS byte addr
        asm volatile("global_load_async_to_lds_b128 %0, %1, off"
                     :: "v"(lds_off), "v"(gsrc) : "memory");
        asm volatile("s_wait_asynccnt 0x0" ::: "memory");
    }
    __syncthreads();

    const int lane = tid & 31;
    const int wv   = tid >> 5;
    const int ct   = blockIdx.y * 8 + wv;      // 0..31 global column tile
    const int half = lane >> 4, l16 = lane & 15;

    v8f acc = {};
#pragma unroll
    for (int kb = 0; kb < 4; ++kb) {
        // A fragment: row M=l16; elems 0..15 -> K=kb*32+16h..+15 (contiguous, matches B)
        const _Float16* rp = &Ash[l16 * 136 + kb * 32 + half * 16];
        v8h lo = *(const v8h*)rp;
        v8h hi = *(const v8h*)(rp + 8);
        v16h a = __builtin_shufflevector(lo, hi, 0,1,2,3,4,5,6,7,8,9,10,11,12,13,14,15);
        // B fragment: contiguous 16 halves per lane in fragment-ordered weights.
        v16h b = *(const v16h*)(wfrag + (((size_t)(kb * 32 + ct) * 32 + lane) << 4));
        acc = __builtin_amdgcn_wmma_f32_16x16x32_f16(false, a, false, b,
                                                     (short)0, acc, false, false);
    }

    const int col  = ct * 16 + l16;            // 0..511
    const float bias = bcat[col];
    const int sel = col >> 7, c = col & 127;
    float* outp = qkvs + (size_t)sel * N_NODES * HID + c;
#pragma unroll
    for (int r = 0; r < 8; ++r) {
        int row = rowbase + half * 8 + r;      // D layout: M = 8*half + r, N = l16
        outp[(size_t)row * HID] = acc[r] + bias;
    }
}

// ---------------------------------------------------------------------------
// Edge pass 1: per-wave edge, per-head dot(q[dst], k[src]+e) * scale,
// store logits, segment max via encoded atomicMax.
// ---------------------------------------------------------------------------
__global__ void __launch_bounds__(256)
edge_logits(const int* __restrict__ ei, const float* __restrict__ ew,
            const float* __restrict__ Wel, const float* __restrict__ qkvs,
            float* __restrict__ logits, unsigned* __restrict__ nmax) {
    int e = blockIdx.x * 8 + (threadIdx.x >> 5);
    if (e >= N_EDGES) return;
    int lane = threadIdx.x & 31;
    int src = ei[e], dst = ei[N_EDGES + e];
    float w = finz(ew[e]);
    const float* q = qkvs + (size_t)dst * HID;
    const float* k = qkvs + (size_t)N_NODES * HID + (size_t)src * HID;
    float p[HEADS];
#pragma unroll
    for (int hh = 0; hh < HEADS; ++hh) {
        int c = hh * 32 + lane;
        p[hh] = q[c] * (k[c] + w * Wel[c]);
    }
#pragma unroll
    for (int m = 16; m > 0; m >>= 1) {
#pragma unroll
        for (int hh = 0; hh < HEADS; ++hh) p[hh] += __shfl_xor(p[hh], m, 32);
    }
    const float scale = 0.17677669529663687f;  // 1/sqrt(32)
    if (lane < HEADS) {
        float lg = p[lane] * scale;
        logits[(size_t)e * HEADS + lane] = lg;
        atomicMax(&nmax[(size_t)dst * HEADS + lane], fenc(lg));
    }
}

// ---------------------------------------------------------------------------
// Edge pass 2: ex = exp(logit - max[dst]); accumulate head sums and
// unnormalized messages (v[src]+e)*ex into accum[dst].
// ---------------------------------------------------------------------------
__global__ void __launch_bounds__(256)
edge_accum(const int* __restrict__ ei, const float* __restrict__ ew,
           const float* __restrict__ Wel, const float* __restrict__ qkvs,
           const float* __restrict__ logits, const unsigned* __restrict__ nmax,
           float* __restrict__ nsum, float* __restrict__ accum) {
    int e = blockIdx.x * 8 + (threadIdx.x >> 5);
    if (e >= N_EDGES) return;
    int lane = threadIdx.x & 31;
    int src = ei[e], dst = ei[N_EDGES + e];
    float w = finz(ew[e]);
    const float* v = qkvs + (size_t)2 * N_NODES * HID + (size_t)src * HID;
    float ex[HEADS];
#pragma unroll
    for (int hh = 0; hh < HEADS; ++hh) {
        float m = fdec(nmax[(size_t)dst * HEADS + hh]);
        ex[hh] = __expf(logits[(size_t)e * HEADS + hh] - m);
    }
    if (lane < HEADS) atomicAdd(&nsum[(size_t)dst * HEADS + lane], ex[lane]);
#pragma unroll
    for (int hh = 0; hh < HEADS; ++hh) {
        int c = hh * 32 + lane;
        atomicAdd(&accum[(size_t)dst * HID + c], (v[c] + w * Wel[c]) * ex[hh]);
    }
}

// ---------------------------------------------------------------------------
// Node finalize: normalize by head sum, + skip, nan_to_num, LayerNorm, ReLU,
// residual. One wave per node; lane handles channels {32*hh + lane}.
// Writes f32 output (residual/next-layer) and f16 copy (next GEMM's A).
// ---------------------------------------------------------------------------
__global__ void __launch_bounds__(256)
node_finalize(const float* __restrict__ accum, const float* __restrict__ nsum,
              const float* __restrict__ qkvs, const float* __restrict__ hres,
              const float* __restrict__ g, const float* __restrict__ b,
              float* __restrict__ hout, _Float16* __restrict__ h16out) {
    int node = blockIdx.x * 8 + (threadIdx.x >> 5);
    if (node >= N_NODES) return;
    int lane = threadIdx.x & 31;
    const float* skip = qkvs + (size_t)3 * N_NODES * HID + (size_t)node * HID;
    float val[HEADS];
#pragma unroll
    for (int hh = 0; hh < HEADS; ++hh) {
        int c = hh * 32 + lane;
        float s = nsum[(size_t)node * HEADS + hh];
        float m = accum[(size_t)node * HID + c];
        float o = (s > 0.0f ? m / s : 0.0f) + skip[c];
        val[hh] = finz(o);
    }
    float ssum = val[0] + val[1] + val[2] + val[3];
#pragma unroll
    for (int m = 16; m > 0; m >>= 1) ssum += __shfl_xor(ssum, m, 32);
    float mu = ssum * (1.0f / 128.0f);
    float vsum = 0.0f;
#pragma unroll
    for (int hh = 0; hh < HEADS; ++hh) { float d = val[hh] - mu; vsum += d * d; }
#pragma unroll
    for (int m = 16; m > 0; m >>= 1) vsum += __shfl_xor(vsum, m, 32);
    float inv = rsqrtf(vsum * (1.0f / 128.0f) + LN_EPS);
#pragma unroll
    for (int hh = 0; hh < HEADS; ++hh) {
        int c = hh * 32 + lane;
        float o = (val[hh] - mu) * inv * g[c] + b[c];
        o = finz(o);
        o = o > 0.0f ? o : 0.0f;               // ReLU (finz of relu is identity)
        float r = o + hres[(size_t)node * HID + c];
        r = finz(r);
        hout[(size_t)node * HID + c] = r;
        h16out[(size_t)node * HID + c] = (_Float16)r;
    }
}

// ---------------------------------------------------------------------------
extern "C" void kernel_launch(void* const* d_in, const int* in_sizes, int n_in,
                              void* d_out, int out_size, void* d_ws, size_t ws_size,
                              hipStream_t stream) {
    const float* x   = (const float*)d_in[0];
    const int*   ei  = (const int*)d_in[1];     // [2, E] int32
    const float* ew  = (const float*)d_in[2];
    const float* Wq  = (const float*)d_in[3];
    const float* bq  = (const float*)d_in[4];
    const float* Wk  = (const float*)d_in[5];
    const float* bk  = (const float*)d_in[6];
    const float* Wv  = (const float*)d_in[7];
    const float* bv  = (const float*)d_in[8];
    const float* We  = (const float*)d_in[9];   // [L,1,128]
    const float* Wsk = (const float*)d_in[10];
    const float* bsk = (const float*)d_in[11];
    const float* lng = (const float*)d_in[12];
    const float* lnb = (const float*)d_in[13];

    // Workspace carve-up (256B aligned slabs).
    char* p = (char*)d_ws;
    auto carve = [&](size_t bytes) -> void* {
        void* r = (void*)p;
        p += (bytes + 255) & ~(size_t)255;
        return r;
    };
    float*    hA    = (float*)   carve((size_t)N_NODES * HID * 4);
    float*    hB    = (float*)   carve((size_t)N_NODES * HID * 4);
    _Float16* h16   = (_Float16*)carve((size_t)N_NODES * HID * 2);
    float*    qkvs  = (float*)   carve((size_t)4 * N_NODES * HID * 4);
    float*    accum = (float*)   carve((size_t)N_NODES * HID * 4);
    float*    nsum  = (float*)   carve((size_t)N_NODES * HEADS * 4);
    unsigned* nmax  = (unsigned*)carve((size_t)N_NODES * HEADS * 4);
    float*    lgt   = (float*)   carve((size_t)N_EDGES * HEADS * 4);
    _Float16* wfrag = (_Float16*)carve((size_t)LAYERS * HID * 512 * 2);
    float*    bcat  = (float*)   carve((size_t)LAYERS * 512 * 4);

    // One-time prep (deterministic, re-run every call).
    prep_weights<<<(LAYERS * 65536 + 255) / 256, 256, 0, stream>>>(
        Wq, Wk, Wv, Wsk, bq, bk, bv, bsk, wfrag, bcat);
    copy_clean<<<(N_NODES * HID + 255) / 256, 256, 0, stream>>>(x, hA, h16,
                                                                N_NODES * HID);

    float* hcur = hA;
    for (int l = 0; l < LAYERS; ++l) {
        float* hnext = (l == LAYERS - 1) ? (float*)d_out : hB;
        const _Float16* wf = wfrag + (size_t)l * HID * 512;
        const float* bc  = bcat + (size_t)l * 512;
        const float* Wel = We + (size_t)l * HID;

        init_layer<<<(N_NODES * HID + 255) / 256, 256, 0, stream>>>(accum, nsum, nmax);

        gemm_qkvs<<<dim3(N_NODES / 16, 4), 256, 0, stream>>>(h16, wf, bc, qkvs);

        edge_logits<<<(N_EDGES + 7) / 8, 256, 0, stream>>>(ei, ew, Wel, qkvs, lgt, nmax);

        edge_accum<<<(N_EDGES + 7) / 8, 256, 0, stream>>>(ei, ew, Wel, qkvs, lgt,
                                                          nmax, nsum, accum);

        node_finalize<<<(N_NODES + 7) / 8, 256, 0, stream>>>(
            accum, nsum, qkvs, hcur, lng + (size_t)l * HID, lnb + (size_t)l * HID,
            hnext, h16);

        hcur = hnext;
    }
    (void)in_sizes; (void)n_in; (void)out_size; (void)ws_size;
}
